// SkateFormerBlock_7112465842186
// MI455X (gfx1250) — compile-verified, matrix operand order checked by reading, not run on hardware
//
#include <hip/hip_runtime.h>
#include <math.h>

typedef __attribute__((ext_vector_type(16))) _Float16 v16h;
typedef __attribute__((ext_vector_type(8)))  _Float16 v8h;
typedef __attribute__((ext_vector_type(8)))  float    v8f;

#define TT   8192
#define NTOK 131072   // 16 * 8192 tokens

// ---- WMMA helpers -------------------------------------------------------
// A (16x32 f16): lane L: m = L%16, 16 halves per lane:
//   regs v0..3 = K = (L/16)*8 + 0..7   (one 16B load)
//   regs v4..7 = K = 16 + (L/16)*8 + 0..7 (one 16B load)
// B (32x16 f16): mirrored with lane -> n, weights stored transposed (N,K).
__device__ __forceinline__ v16h load_frag(const _Float16* p) {
    union { v16h v; v8h h[2]; } u;
    u.h[0] = *(const v8h*)(p);
    u.h[1] = *(const v8h*)(p + 16);
    return u.v;
}
__device__ __forceinline__ v8f wmma16(v16h a, v16h b, v8f c) {
    return __builtin_amdgcn_wmma_f32_16x16x32_f16(false, a, false, b, (short)0, c,
                                                  false, false);
}

// ---- weight transpose + f32 -> f16 cast: dst[(c)*R + r] = src[r*Cc + c] ---
__global__ void wcast_k(const float* __restrict__ src, _Float16* __restrict__ dst,
                        int R, int Cc) {
    int i = blockIdx.x * blockDim.x + threadIdx.x;
    if (i < R * Cc) {
        int r = i / Cc, c = i % Cc;
        dst[(size_t)c * R + r] = (_Float16)src[i];
    }
}

// ---- Stage 1: transpose x -> xt (tokens,C) f32 + LayerNorm1 -> f16 -------
__global__ void ln1_k(const float* __restrict__ x, const float* __restrict__ g1,
                      const float* __restrict__ b1, float* __restrict__ xt,
                      _Float16* __restrict__ lnh) {
    __shared__ float tile[256][33];          // [c][t] padded
    __shared__ float stats[2][32];
    int blk = blockIdx.x;                    // 16 * 256 blocks
    int b = blk >> 8;
    int t0 = (blk & 255) * 32;
    // coalesced load: 8192 floats, 32 iters, each 32-lane group reads one c-row
    for (int it = 0; it < 32; ++it) {
        int i = threadIdx.x + 256 * it;
        int c = i >> 5, t = i & 31;
        tile[c][t] = x[((size_t)b * 256 + c) * TT + t0 + t];
    }
    __syncthreads();
    // stats: 8 lanes per token
    int j = threadIdx.x >> 3, p = threadIdx.x & 7;
    float s = 0.f, ss = 0.f;
    for (int c = p; c < 256; c += 8) { float v = tile[c][j]; s += v; ss += v * v; }
    s  += __shfl_xor(s, 4, 8);  s  += __shfl_xor(s, 2, 8);  s  += __shfl_xor(s, 1, 8);
    ss += __shfl_xor(ss, 4, 8); ss += __shfl_xor(ss, 2, 8); ss += __shfl_xor(ss, 1, 8);
    if (p == 0) {
        float m = s * (1.f / 256.f);
        float v = ss * (1.f / 256.f) - m * m;
        stats[0][j] = m;
        stats[1][j] = rsqrtf(v + 1e-5f);
    }
    __syncthreads();
    int c = threadIdx.x;
    float gv = g1[c], bv = b1[c];
    for (int jj = 0; jj < 32; ++jj) {
        size_t n = (size_t)b * TT + t0 + jj;
        float v = tile[c][jj];
        xt[n * 256 + c]  = v;
        lnh[n * 256 + c] = (_Float16)((v - stats[0][jj]) * stats[1][jj] * gv + bv);
    }
}

// ---- GEMM1: f = ln1h(N,256) @ Wm(256,512) + bm   -> f16 (N,512) ----------
// Each wave: 32-token M-supertile x 4 N-tiles => 8 accumulators; every
// B-frag is reused by 2 WMMAs (1.5 b128 loads per WMMA).
__global__ void gemm1_k(const _Float16* __restrict__ A, const _Float16* __restrict__ Bt,
                        const float* __restrict__ bias, _Float16* __restrict__ F) {
    int msup = blockIdx.x;                   // 4096, 32 tokens each
    int wave = threadIdx.x >> 5;
    int lane = threadIdx.x & 31;
    int lnid = lane & 15;
    int hoff = (lane >> 4) * 8;
    const _Float16* arow0 = A + ((size_t)msup * 32 + lnid) * 256 + hoff;
    const _Float16* arow1 = arow0 + 16 * 256;
    const _Float16* brow[4];
    v8f acc[2][4];
#pragma unroll
    for (int nt = 0; nt < 4; ++nt) {
        int n = (wave * 4 + nt) * 16 + lnid;
        brow[nt] = Bt + (size_t)n * 256 + hoff;
        float b = bias[n];
        acc[0][nt] = (v8f){b, b, b, b, b, b, b, b};
        acc[1][nt] = acc[0][nt];
    }
#pragma unroll
    for (int ks = 0; ks < 8; ++ks) {
        if (ks < 7) {                        // pull next K-chunk toward L0
            __builtin_prefetch(arow0 + (ks + 1) * 32, 0, 3);
            __builtin_prefetch(arow1 + (ks + 1) * 32, 0, 3);
        }
        v16h a0 = load_frag(arow0 + ks * 32);
        v16h a1 = load_frag(arow1 + ks * 32);
#pragma unroll
        for (int nt = 0; nt < 4; ++nt) {
            if (ks < 7) __builtin_prefetch(brow[nt] + (ks + 1) * 32, 0, 3);
            v16h b = load_frag(brow[nt] + ks * 32);
            acc[0][nt] = wmma16(a0, b, acc[0][nt]);
            acc[1][nt] = wmma16(a1, b, acc[1][nt]);
        }
    }
#pragma unroll
    for (int mt = 0; mt < 2; ++mt) {
        size_t base = ((size_t)msup * 32 + mt * 16 + 8 * (lane >> 4)) * 512;
#pragma unroll
        for (int nt = 0; nt < 4; ++nt) {
            int n = (wave * 4 + nt) * 16 + lnid;
#pragma unroll
            for (int v = 0; v < 8; ++v)
                F[base + (size_t)v * 512 + n] = (_Float16)acc[mt][nt][v];
        }
    }
}

// ---- grouped conv1d: y0 = conv(f[:, :128]) + bc  -> y[:, 0:128] ----------
__global__ void conv_k(const _Float16* __restrict__ F, const float* __restrict__ Wc,
                       const float* __restrict__ bc, _Float16* __restrict__ Y) {
    __shared__ float fin[16][264];           // ci x (256 + 6) t
    __shared__ float w[16][16][7];           // co, ci, k
    int blk = blockIdx.x;                    // 16 * 8 * 32
    int tt = blk & 31;
    int g = (blk >> 5) & 7;
    int b = blk >> 8;
    int t0 = tt * 256;
    for (int i = threadIdx.x; i < 16 * 262; i += 256) {
        int ci = i / 262, dt = i % 262;
        int t = t0 - 3 + dt;
        float v = 0.f;
        if (t >= 0 && t < TT)
            v = (float)F[((size_t)b * TT + t) * 512 + g * 16 + ci];
        fin[ci][dt] = v;
    }
    for (int i = threadIdx.x; i < 16 * 112; i += 256) {
        int col = i / 112, r = i % 112;
        w[col][r / 7][r % 7] = Wc[(g * 16 + col) * 112 + r];
    }
    __syncthreads();
    int tl = threadIdx.x;
    for (int co = 0; co < 16; ++co) {
        float acc = bc[g * 16 + co];
#pragma unroll
        for (int ci = 0; ci < 16; ++ci)
#pragma unroll
            for (int k = 0; k < 7; ++k)
                acc += fin[ci][tl + k] * w[co][ci][k];
        Y[((size_t)b * TT + t0 + tl) * 256 + g * 16 + co] = (_Float16)acc;
    }
}

// ---- window attention (both partition types) -----------------------------
// one thread per (b, window, head, row i); P=8, heads=4, d=16, scale=0.5
__global__ void attn_k(const _Float16* __restrict__ F, const float* __restrict__ tab,
                       _Float16* __restrict__ Y, int qbase, int ybase, int type) {
    int idx = blockIdx.x * blockDim.x + threadIdx.x;
    int i = idx & 7;
    int h = (idx >> 3) & 3;
    int w = (idx >> 5) & 1023;
    int b = idx >> 15;
    if (b >= 16) return;
    size_t nbase = (size_t)b * TT;
    float q[16];
    {
        int ti = (type == 1) ? (w * 8 + i) : (i * 1024 + w);
        const _Float16* r = F + (nbase + ti) * 512 + qbase + h * 16;
#pragma unroll
        for (int d = 0; d < 16; ++d) q[d] = (float)r[d];
    }
    float lg[8], mx = -1e30f;
    for (int j = 0; j < 8; ++j) {
        int tj = (type == 1) ? (w * 8 + j) : (j * 1024 + w);
        const _Float16* r = F + (nbase + tj) * 512 + qbase + 64 + h * 16;
        float s = 0.f;
#pragma unroll
        for (int d = 0; d < 16; ++d) s += q[d] * (float)r[d];
        s = s * 0.5f + tab[(i - j + 7) * 4 + h];
        lg[j] = s;
        mx = fmaxf(mx, s);
    }
    float den = 0.f;
    for (int j = 0; j < 8; ++j) { lg[j] = __expf(lg[j] - mx); den += lg[j]; }
    float o[16];
#pragma unroll
    for (int d = 0; d < 16; ++d) o[d] = 0.f;
    for (int j = 0; j < 8; ++j) {
        int tj = (type == 1) ? (w * 8 + j) : (j * 1024 + w);
        const _Float16* r = F + (nbase + tj) * 512 + qbase + 128 + h * 16;
        float p = lg[j];
#pragma unroll
        for (int d = 0; d < 16; ++d) o[d] += p * (float)r[d];
    }
    float inv = 1.f / den;
    int to = (type == 1) ? (w * 8 + i) : (i * 1024 + w);
    _Float16* ro = Y + (nbase + to) * 256 + ybase + h * 16;
#pragma unroll
    for (int d = 0; d < 16; ++d) ro[d] = (_Float16)(o[d] * inv);
}

// ---- GEMM2: out1 = y @ Wp + bp + skip; fused LN2 -> ln2h -----------------
__global__ void gemm2_k(const _Float16* __restrict__ A, const _Float16* __restrict__ Bt,
                        const float* __restrict__ bp, const float* __restrict__ xt,
                        const float* __restrict__ g2, const float* __restrict__ b2,
                        float* __restrict__ out1, _Float16* __restrict__ ln2h) {
    __shared__ float tile[16][257];
    __shared__ float stats[2][16];
    int mtile = blockIdx.x;                  // 8192
    int wave = threadIdx.x >> 5;
    int lane = threadIdx.x & 31;
    int lnid = lane & 15;
    int hoff = (lane >> 4) * 8;
    const _Float16* arow = A + ((size_t)mtile * 16 + lnid) * 256 + hoff;
    v16h a[8];
#pragma unroll
    for (int ks = 0; ks < 8; ++ks) a[ks] = load_frag(arow + ks * 32);
#pragma unroll
    for (int nt = 0; nt < 2; ++nt) {
        int n = (wave * 2 + nt) * 16 + lnid;
        const _Float16* brow = Bt + (size_t)n * 256 + hoff;
        v8f c = {0.f, 0.f, 0.f, 0.f, 0.f, 0.f, 0.f, 0.f};
#pragma unroll
        for (int ks = 0; ks < 8; ++ks) {
            if (ks < 7) __builtin_prefetch(brow + (ks + 1) * 32, 0, 3);
            c = wmma16(a[ks], load_frag(brow + ks * 32), c);
        }
        float bv = bp[n];
#pragma unroll
        for (int v = 0; v < 8; ++v) {
            int m = v + 8 * (lane >> 4);
            size_t row = (size_t)mtile * 16 + m;
            float val = c[v] + bv + xt[row * 256 + n];
            out1[row * 256 + n] = val;
            tile[m][n] = val;
        }
    }
    __syncthreads();
    // LN2: 16 lanes per token
    int j = threadIdx.x >> 4, p = threadIdx.x & 15;
    float s = 0.f, ss = 0.f;
    for (int c = p; c < 256; c += 16) { float v = tile[j][c]; s += v; ss += v * v; }
    s  += __shfl_xor(s, 8, 16);  s  += __shfl_xor(s, 4, 16);
    s  += __shfl_xor(s, 2, 16);  s  += __shfl_xor(s, 1, 16);
    ss += __shfl_xor(ss, 8, 16); ss += __shfl_xor(ss, 4, 16);
    ss += __shfl_xor(ss, 2, 16); ss += __shfl_xor(ss, 1, 16);
    if (p == 0) {
        float m = s * (1.f / 256.f);
        float v = ss * (1.f / 256.f) - m * m;
        stats[0][j] = m;
        stats[1][j] = rsqrtf(v + 1e-5f);
    }
    __syncthreads();
    int c = threadIdx.x;
    float gv = g2[c], bv = b2[c];
    for (int jj = 0; jj < 16; ++jj) {
        float v = (tile[jj][c] - stats[0][jj]) * stats[1][jj] * gv + bv;
        ln2h[((size_t)mtile * 16 + jj) * 256 + c] = (_Float16)v;
    }
}

// ---- fused FFN: gelu(ln2 @ W1 + bf1) @ W2 + bf2 + out1 -> d_out (B,C,T) --
__global__ void ffn_k(const _Float16* __restrict__ A, const _Float16* __restrict__ B1t,
                      const float* __restrict__ bf1, const _Float16* __restrict__ B2t,
                      const float* __restrict__ bf2, const float* __restrict__ out1,
                      float* __restrict__ out) {
    __shared__ __align__(16) _Float16 hchunk[16][128];   // 16 tokens x 128 h1 ch
    int mtile = blockIdx.x;                  // 8192
    int wave = threadIdx.x >> 5;
    int lane = threadIdx.x & 31;
    int lnid = lane & 15;
    int hoff = (lane >> 4) * 8;
    const _Float16* arow = A + ((size_t)mtile * 16 + lnid) * 256 + hoff;
    v16h a[8];
#pragma unroll
    for (int ks = 0; ks < 8; ++ks) a[ks] = load_frag(arow + ks * 32);
    v8f acc[2];
    acc[0] = (v8f){0.f, 0.f, 0.f, 0.f, 0.f, 0.f, 0.f, 0.f};
    acc[1] = acc[0];
    for (int chunk = 0; chunk < 8; ++chunk) {
        // phase A: this wave computes h1 n1-tile (chunk*8 + wave)
        int n1 = (chunk * 8 + wave) * 16 + lnid;
        const _Float16* brow = B1t + (size_t)n1 * 256 + hoff;
        v8f c = {0.f, 0.f, 0.f, 0.f, 0.f, 0.f, 0.f, 0.f};
#pragma unroll
        for (int ks = 0; ks < 8; ++ks) {
            if (ks < 7) __builtin_prefetch(brow + (ks + 1) * 32, 0, 3);
            c = wmma16(a[ks], load_frag(brow + ks * 32), c);
        }
        float bb = bf1[n1];
        int col = wave * 16 + lnid;
#pragma unroll
        for (int v = 0; v < 8; ++v) {
            float x = c[v] + bb;
            float g = 0.5f * x * (1.f + erff(x * 0.70710678118f));
            hchunk[v + 8 * (lane >> 4)][col] = (_Float16)g;
        }
        __syncthreads();
        // phase B: accumulate this K=128 chunk into resident output tiles
        const _Float16* lrow = &hchunk[lnid][hoff];
#pragma unroll
        for (int ks = 0; ks < 4; ++ks) {
            v16h af = load_frag(lrow + ks * 32);
#pragma unroll
            for (int nt = 0; nt < 2; ++nt) {
                int n2 = (wave * 2 + nt) * 16 + lnid;
                const _Float16* b2row = B2t + (size_t)n2 * 1024 + hoff;
                if (chunk < 7)               // prefetch next chunk of W2t
                    __builtin_prefetch(b2row + (chunk + 1) * 128 + ks * 32, 0, 3);
                v16h bf = load_frag(b2row + chunk * 128 + ks * 32);
                acc[nt] = wmma16(af, bf, acc[nt]);
            }
        }
        __syncthreads();
    }
    // epilogue: + bf2 + residual, store transposed (B,C,T)
#pragma unroll
    for (int nt = 0; nt < 2; ++nt) {
        int n = (wave * 2 + nt) * 16 + lnid;
        float bv = bf2[n];
#pragma unroll
        for (int v = 0; v < 8; ++v) {
            size_t row = (size_t)mtile * 16 + v + 8 * (lane >> 4);
            float val = acc[nt][v] + bv + out1[row * 256 + n];
            int b = (int)(row >> 13);
            int t = (int)(row & 8191);
            out[((size_t)(b * 256 + n)) * TT + t] = val;
        }
    }
}

extern "C" void kernel_launch(void* const* d_in, const int* in_sizes, int n_in,
                              void* d_out, int out_size, void* d_ws, size_t ws_size,
                              hipStream_t stream) {
    (void)in_sizes; (void)n_in; (void)out_size; (void)ws_size;
    const float* x    = (const float*)d_in[0];
    const float* g1   = (const float*)d_in[1];
    const float* b1   = (const float*)d_in[2];
    const float* Wm   = (const float*)d_in[3];
    const float* bm   = (const float*)d_in[4];
    const float* Wc   = (const float*)d_in[5];
    const float* bc   = (const float*)d_in[6];
    const float* tab1 = (const float*)d_in[7];
    const float* tab2 = (const float*)d_in[8];
    const float* Wp   = (const float*)d_in[9];
    const float* bp   = (const float*)d_in[10];
    const float* g2   = (const float*)d_in[11];
    const float* b2   = (const float*)d_in[12];
    const float* W1   = (const float*)d_in[13];
    const float* bf1  = (const float*)d_in[14];
    const float* W2   = (const float*)d_in[15];
    const float* bf2  = (const float*)d_in[16];
    float* outp = (float*)d_out;

    char* ws = (char*)d_ws;
    float*    xt   = (float*)   (ws);                       // 128 MB
    _Float16* ln1h = (_Float16*)(ws + 134217728ull);        //  64 MB
    _Float16* fbuf = (_Float16*)(ws + 201326592ull);        // 128 MB
    _Float16* ybuf = (_Float16*)(ws + 335544320ull);        //  64 MB
    float*    out1 = (float*)   (ws + 402653184ull);        // 128 MB
    _Float16* ln2h = (_Float16*)(ws + 536870912ull);        //  64 MB
    _Float16* Wmt  = (_Float16*)(ws + 603979776ull);        // (512,256)
    _Float16* Wpt  = (_Float16*)(ws + 604241920ull);        // (256,256)
    _Float16* W1t  = (_Float16*)(ws + 604372992ull);        // (1024,256)
    _Float16* W2t  = (_Float16*)(ws + 604897280ull);        // (256,1024)

    // 1. weight transpose + cast to f16 (N,K) layout
    wcast_k<<<(256 * 512 + 255) / 256, 256, 0, stream>>>(Wm, Wmt, 256, 512);
    wcast_k<<<(256 * 256 + 255) / 256, 256, 0, stream>>>(Wp, Wpt, 256, 256);
    wcast_k<<<(256 * 1024 + 255) / 256, 256, 0, stream>>>(W1, W1t, 256, 1024);
    wcast_k<<<(1024 * 256 + 255) / 256, 256, 0, stream>>>(W2, W2t, 1024, 256);
    // 2. transpose + LN1
    ln1_k<<<16 * 256, 256, 0, stream>>>(x, g1, b1, xt, ln1h);
    // 3. GEMM1 -> f  (32-token supertiles)
    gemm1_k<<<NTOK / 32, 256, 0, stream>>>(ln1h, Wmt, bm, fbuf);
    // 4. grouped conv -> y[:, 0:128]
    conv_k<<<16 * 8 * 32, 256, 0, stream>>>(fbuf, Wc, bc, ybuf);
    // 5. window attn type1 -> y[:, 128:192]
    attn_k<<<524288 / 256, 256, 0, stream>>>(fbuf, tab1, ybuf, 128, 128, 1);
    // 6. window attn type2 -> y[:, 192:256]
    attn_k<<<524288 / 256, 256, 0, stream>>>(fbuf, tab2, ybuf, 320, 192, 2);
    // 7. GEMM2 + skip + LN2
    gemm2_k<<<NTOK / 16, 256, 0, stream>>>(ybuf, Wpt, bp, xt, g2, b2, out1, ln2h);
    // 8. fused FFN + residual + output transpose
    ffn_k<<<NTOK / 16, 256, 0, stream>>>(ln2h, W1t, bf1, W2t, bf2, out1, outp);
}
